// rago_55319178772907
// MI455X (gfx1250) — matrix operand order repaired and checked by reading, not compiled, not run on hardware
//
// RAGO rotation-averaging GNN forward for MI455X (gfx1250, wave32, WMMA).
// All dense layers run through v_wmma_f32_16x16x32_bf16 with LDS-staged
// bf16 tiles; scatter means via global_atomic_add_f32; geometry in fp32 VALU.
#include <hip/hip_runtime.h>
#include <hip/hip_bf16.h>
#include <math.h>

#define HDIM 48
#define KMAX 160   // max K is 144 (H + 2H concats), padded to 32

typedef __attribute__((ext_vector_type(16))) __bf16 bf16x16;
typedef __attribute__((ext_vector_type(8)))  float  f32x8;

enum { ACT_NONE = 0, ACT_RELU = 1, ACT_TANH = 2, ACT_SIG = 3 };

__device__ __forceinline__ __bf16 f2bf(float f) {
  unsigned u = __builtin_bit_cast(unsigned, f);
  unsigned short h = (unsigned short)((u + 0x7FFFu + ((u >> 16) & 1u)) >> 16);
  return __builtin_bit_cast(__bf16, h);
}

// ---------------------------------------------------------------------------
// Generic fused GEMM: Out[M x N] = act( concat(A0|A1|A2) @ W[KxN] + bias )
// Each segment may gather rows through an int index array (for x[ei0] etc).
// Block = 128 threads = 4 waves; each wave owns a 16-row stripe of a 64-row
// block tile, looping over 16-col tiles of N (N <= 48) with K tiled by 32.
// A tile and transposed weight panel live in LDS as bf16; WMMA operands are
// built with dword ds loads following the CDNA5 16-bit A/B lane striping.
// ---------------------------------------------------------------------------
__global__ __launch_bounds__(128)
void rago_gemm(const float* __restrict__ A0, int w0, const int* __restrict__ g0,
               const float* __restrict__ A1, int w1, const int* __restrict__ g1,
               const float* __restrict__ A2, int w2, const int* __restrict__ g2,
               const float* __restrict__ W, const float* __restrict__ bias,
               float* __restrict__ Out, int M, int N, int K, int act) {
  __shared__ __bf16 lA[64][KMAX];    // activation tile, row-major
  __shared__ __bf16 lB[HDIM][KMAX];  // weights transposed: lB[n][k]
  const int Kpad = (K + 31) & ~31;
  const int bm = blockIdx.x * 64;

  // Stage A: gather + convert fp32 -> bf16 (zero pad K..Kpad and rows >= M)
  for (int t = threadIdx.x; t < 64 * Kpad; t += 128) {
    int lr = t / Kpad, c = t - lr * Kpad;
    int r = bm + lr;
    float v = 0.f;
    if (r < M && c < K) {
      if (c < w0) {
        int rr = g0 ? g0[r] : r;
        v = A0[(size_t)rr * w0 + c];
      } else if (c < w0 + w1) {
        int rr = g1 ? g1[r] : r;
        v = A1[(size_t)rr * w1 + (c - w0)];
      } else {
        int rr = g2 ? g2[r] : r;
        v = A2[(size_t)rr * w2 + (c - w0 - w1)];
      }
    }
    lA[lr][c] = f2bf(v);
  }
  // Stage B transposed (weights are tiny and L2-hot; cols >= N zero-filled)
  for (int t = threadIdx.x; t < HDIM * Kpad; t += 128) {
    int n = t / Kpad, kk = t - n * Kpad;
    float v = (kk < K && n < N) ? W[(size_t)kk * N + n] : 0.f;
    lB[n][kk] = f2bf(v);
  }
  __syncthreads();

  const int wv = threadIdx.x >> 5;
  const int lane = threadIdx.x & 31;
  const int half = lane >> 4;   // which 16-lane half of the wave
  const int l16 = lane & 15;
  const int mloc = wv * 16 + l16;        // A row this lane feeds
  const int NT = (N + 15) >> 4;

  union AB { bf16x16 v; unsigned u[8]; };

  for (int nt = 0; nt < NT; ++nt) {
    const int ncol = nt * 16 + l16;      // < 48 always; zero cols beyond N
    f32x8 acc = {0.f, 0.f, 0.f, 0.f, 0.f, 0.f, 0.f, 0.f};
    for (int kt = 0; kt < Kpad; kt += 32) {
      AB a, b;
#pragma unroll
      for (int v = 0; v < 8; ++v) {
        // A 16x32 bf16 striping: VGPR v<4 -> K=2v(+half*8); v>=4 -> 16+2(v-4)(+half*8)
        int ka = kt + ((v < 4) ? (2 * v) : (16 + 2 * (v - 4))) + half * 8;
        // B 32x16 bf16 striping: same K order as A: VGPR v<4 -> K=2v(+half*8); v>=4 -> 16+2(v-4)(+half*8)
        int kb = kt + ((v < 4) ? (2 * v) : (16 + 2 * (v - 4))) + half * 8;
        a.u[v] = *(const unsigned*)(&lA[mloc][ka]);
        b.u[v] = *(const unsigned*)(&lB[ncol][kb]);
      }
      acc = __builtin_amdgcn_wmma_f32_16x16x32_bf16(false, a.v, false, b.v,
                                                    (short)0, acc, false, false);
    }
#pragma unroll
    for (int r = 0; r < 8; ++r) {
      int mg = bm + wv * 16 + half * 8 + r;  // C/D layout: VGPR r -> M=r(+8)
      if (mg < M && ncol < N) {
        float vv = acc[r] + (bias ? bias[ncol] : 0.f);
        if (act == ACT_RELU) vv = fmaxf(vv, 0.f);
        else if (act == ACT_TANH) vv = tanhf(vv);
        else if (act == ACT_SIG) vv = 1.f / (1.f + __expf(-vv));
        Out[(size_t)mg * N + ncol] = vv;
      }
    }
  }
}

// ---------------------------------------------------------------------------
// ortho6d: 6D -> 3x3 rotation, stored stack([b1,b2,b3], axis=-1) row-major
// ---------------------------------------------------------------------------
__device__ __forceinline__ void rago_ortho6d(const float* __restrict__ x,
                                             float* __restrict__ m) {
  float n1 = sqrtf(x[0]*x[0] + x[1]*x[1] + x[2]*x[2]) + 1e-8f;
  float b1x = x[0]/n1, b1y = x[1]/n1, b1z = x[2]/n1;
  float cx = b1y*x[5] - b1z*x[4];
  float cy = b1z*x[3] - b1x*x[5];
  float cz = b1x*x[4] - b1y*x[3];
  float n3 = sqrtf(cx*cx + cy*cy + cz*cz) + 1e-8f;
  float b3x = cx/n3, b3y = cy/n3, b3z = cz/n3;
  float b2x = b3y*b1z - b3z*b1y;
  float b2y = b3z*b1x - b3x*b1z;
  float b2z = b3x*b1y - b3y*b1x;
  m[0]=b1x; m[1]=b2x; m[2]=b3x;
  m[3]=b1y; m[4]=b2y; m[5]=b3y;
  m[6]=b1z; m[7]=b2z; m[8]=b3z;
}

// graph_cost: per edge, |EA^T R_j - R_j| scattered (atomic) into segsum/cnt,
// and ecost = |ortho6d(edge_rot6) - EA| + |ortho6d(edge_rot6) - R_i^T R_j|
__global__ void rago_graph_cost(const float* __restrict__ node_rot,
                                const float* __restrict__ edge_rot6,
                                const float* __restrict__ edge_attr,
                                const int* __restrict__ ei0,
                                const int* __restrict__ ei1,
                                float* __restrict__ ecost,
                                float* __restrict__ segsum,
                                float* __restrict__ segcnt, int E) {
  int e = blockIdx.x * blockDim.x + threadIdx.x;
  if (e >= E) return;
  int i = ei0[e], j = ei1[e];
  float EA[9], EE[9], RI[9];
#pragma unroll
  for (int t = 0; t < 9; ++t) {
    EA[t] = edge_attr[(size_t)e * 9 + t];
    EE[t] = node_rot[(size_t)j * 9 + t];
    RI[t] = node_rot[(size_t)i * 9 + t];
  }
#pragma unroll
  for (int a = 0; a < 3; ++a)
#pragma unroll
    for (int b = 0; b < 3; ++b) {
      float s = EA[a]*EE[b] + EA[3+a]*EE[3+b] + EA[6+a]*EE[6+b]; // (EA^T EE)ab
      atomicAdd(&segsum[(size_t)j * 9 + a*3 + b], fabsf(s - EE[a*3 + b]));
    }
  atomicAdd(&segcnt[j], 1.f);
  float erm[9];
  rago_ortho6d(&edge_rot6[(size_t)e * 6], erm);
#pragma unroll
  for (int a = 0; a < 3; ++a)
#pragma unroll
    for (int b = 0; b < 3; ++b) {
      float nr = RI[a]*EE[b] + RI[3+a]*EE[3+b] + RI[6+a]*EE[6+b]; // (R_i^T R_j)ab
      int o = a*3 + b;
      ecost[(size_t)e * 9 + o] = fabsf(erm[o] - EA[o]) + fabsf(erm[o] - nr);
    }
}

// node_rot = node_rot @ ortho6d(d6); est = ortho6d([col0(node_rot), col1])
__global__ void rago_node_update(float* __restrict__ node_rot,
                                 const float* __restrict__ d6,
                                 float* __restrict__ est, int Nn) {
  int n = blockIdx.x * blockDim.x + threadIdx.x;
  if (n >= Nn) return;
  float R[9], dR[9], Rn[9];
#pragma unroll
  for (int t = 0; t < 9; ++t) R[t] = node_rot[(size_t)n * 9 + t];
  rago_ortho6d(&d6[(size_t)n * 6], dR);
#pragma unroll
  for (int a = 0; a < 3; ++a)
#pragma unroll
    for (int b = 0; b < 3; ++b)
      Rn[a*3+b] = R[a*3]*dR[b] + R[a*3+1]*dR[3+b] + R[a*3+2]*dR[6+b];
#pragma unroll
  for (int t = 0; t < 9; ++t) node_rot[(size_t)n * 9 + t] = Rn[t];
  float six[6] = {Rn[0], Rn[3], Rn[6], Rn[1], Rn[4], Rn[7]};
  float em[9];
  rago_ortho6d(six, em);
#pragma unroll
  for (int t = 0; t < 9; ++t) est[(size_t)n * 9 + t] = em[t];
}

// ------------------------- elementwise helpers -----------------------------
__global__ void rago_fill0(float* __restrict__ p, size_t n) {
  size_t i = (size_t)blockIdx.x * blockDim.x + threadIdx.x;
  if (i < n) p[i] = 0.f;
}
__global__ void rago_tanh_ip(float* __restrict__ p, size_t n) {
  size_t i = (size_t)blockIdx.x * blockDim.x + threadIdx.x;
  if (i < n) p[i] = tanhf(p[i]);
}
__global__ void rago_mul(const float* __restrict__ a, const float* __restrict__ b,
                         float* __restrict__ o, size_t n) {
  size_t i = (size_t)blockIdx.x * blockDim.x + threadIdx.x;
  if (i < n) o[i] = a[i] * b[i];
}
__global__ void rago_gru_combine(float* __restrict__ h, const float* __restrict__ z,
                                 const float* __restrict__ q, size_t n) {
  size_t i = (size_t)blockIdx.x * blockDim.x + threadIdx.x;
  if (i < n) {
    float zz = z[i];
    h[i] = tanhf((1.f - zz) * h[i] + zz * q[i]);  // tanh(gru(...)) fused
  }
}
__global__ void rago_edge_update(float* __restrict__ er, const float* __restrict__ d,
                                 float* __restrict__ rect, size_t n) {
  size_t i = (size_t)blockIdx.x * blockDim.x + threadIdx.x;
  if (i < n) {
    float v = er[i] + d[i];
    er[i] = v;
    rect[i] = v;
  }
}
__global__ void rago_seg_accum(const float* __restrict__ vals, const int* __restrict__ idx,
                               float* __restrict__ sum, float* __restrict__ cnt,
                               int E, int w) {
  size_t i = (size_t)blockIdx.x * blockDim.x + threadIdx.x;
  size_t tot = (size_t)E * w;
  if (i >= tot) return;
  int e = (int)(i / w);
  int c = (int)(i - (size_t)e * w);
  int s = idx[e];
  atomicAdd(&sum[(size_t)s * w + c], vals[i]);
  if (c == 0) atomicAdd(&cnt[s], 1.f);
}
__global__ void rago_seg_div(const float* __restrict__ sum, const float* __restrict__ cnt,
                             float* __restrict__ out, int n, int w) {
  size_t i = (size_t)blockIdx.x * blockDim.x + threadIdx.x;
  if (i >= (size_t)n * w) return;
  out[i] = sum[i] / fmaxf(cnt[i / w], 1.f);
}

// ---------------------------------------------------------------------------
extern "C" void kernel_launch(void* const* d_in, const int* in_sizes, int n_in,
                              void* d_out, int out_size, void* d_ws, size_t ws_size,
                              hipStream_t stream) {
  const int Nn = in_sizes[0] / 9;
  const int E  = in_sizes[1] / 9;
  const float* init_rot  = (const float*)d_in[0];
  const float* edge_attr = (const float*)d_in[1];
  const int* edge_index  = (const int*)d_in[n_in - 1];
  const int* ei0 = edge_index;
  const int* ei1 = edge_index + E;

  // Params in setup_inputs() insertion order, flattened recursively.
  int pi = 2;
  auto P = [&]() { return (const float*)d_in[pi++]; };
  const float *MF_We, *MF_be, *MF_Wn, *MF_bn;
  const float *MS_We, *MS_be, *MS_Wn, *MS_bn;
  const float *MEC_We[3], *MEC_be[3], *MEC_Wn[3], *MEC_bn[3];
  const float *MNC_We[3], *MNC_be[3], *MNC_Wn[3], *MNC_bn[3];
  const float *EF_W[6], *EF_b[6], *EU_W[6], *EU_b[6];
  const float *NF_W[6], *NF_b[6], *NU_W[6], *NU_b[6];
  const float *EG_Wz, *EG_bz, *EG_Wr, *EG_br, *EG_Wq, *EG_bq;
  const float *NG_Wz, *NG_bz, *NG_Wr, *NG_br, *NG_Wq, *NG_bq;
  MF_We = P(); MF_be = P(); MF_Wn = P(); MF_bn = P();              // mpnn_feat
  MS_We = P(); MS_be = P(); MS_Wn = P(); MS_bn = P();              // mpnn_state
  for (int i = 0; i < 3; ++i) { MEC_We[i]=P(); MEC_be[i]=P(); MEC_Wn[i]=P(); MEC_bn[i]=P(); }
  for (int i = 0; i < 3; ++i) { MNC_We[i]=P(); MNC_be[i]=P(); MNC_Wn[i]=P(); MNC_bn[i]=P(); }
  for (int i = 0; i < 6; ++i) { EF_W[i]=P(); EF_b[i]=P(); }        // edge_fusion
  EG_Wz=P(); EG_bz=P(); EG_Wr=P(); EG_br=P(); EG_Wq=P(); EG_bq=P();// edge_gru
  for (int i = 0; i < 6; ++i) { EU_W[i]=P(); EU_b[i]=P(); }        // edge_updater
  for (int i = 0; i < 6; ++i) { NF_W[i]=P(); NF_b[i]=P(); }        // node_fusion
  NG_Wz=P(); NG_bz=P(); NG_Wr=P(); NG_br=P(); NG_Wq=P(); NG_bq=P();// node_gru
  for (int i = 0; i < 6; ++i) { NU_W[i]=P(); NU_b[i]=P(); }        // node_updater

  // Workspace carve (bump allocator, 256B aligned)
  char* wp = (char*)d_ws;
  auto alloc = [&](size_t elems) {
    float* r = (float*)wp;
    wp += ((elems * sizeof(float)) + 255) & ~(size_t)255;
    return r;
  };
  float* node_rot   = alloc((size_t)Nn * 9);
  float* edge_rot   = alloc((size_t)E * 6);
  float* node_feat  = alloc((size_t)Nn * HDIM);
  float* edge_feat  = alloc((size_t)E * HDIM);
  float* node_state = alloc((size_t)Nn * HDIM);
  float* edge_state = alloc((size_t)E * HDIM);
  float* segsum     = alloc((size_t)Nn * HDIM);
  float* segcnt     = alloc((size_t)Nn);
  float* t0 = alloc((size_t)E * HDIM);
  float* t1 = alloc((size_t)E * HDIM);
  float* t2 = alloc((size_t)E * HDIM);
  float* t3 = alloc((size_t)E * HDIM);
  float* t4 = alloc((size_t)E * HDIM);
  float* n0 = alloc((size_t)Nn * HDIM);
  float* n1 = alloc((size_t)Nn * HDIM);
  float* n2 = alloc((size_t)Nn * HDIM);
  float* n3 = alloc((size_t)Nn * HDIM);
  float* n4 = alloc((size_t)Nn * HDIM);
  (void)ws_size; (void)out_size;

  auto ewg = [](size_t n) { return dim3((unsigned)((n + 255) / 256)); };
  auto gemm = [&](const float* A0, int w0, const int* g0,
                  const float* A1, int w1, const int* g1,
                  const float* A2, int w2, const int* g2,
                  const float* W, const float* b, float* Out,
                  int M, int Ncols, int act) {
    dim3 g((unsigned)((M + 63) / 64));
    rago_gemm<<<g, 128, 0, stream>>>(A0, w0, g0, A1, w1, g1, A2, w2, g2,
                                     W, b, Out, M, Ncols, w0 + w1 + w2, act);
  };
  auto segmean = [&](const float* vals, int w, float* out) {
    rago_fill0<<<ewg((size_t)Nn * w), 256, 0, stream>>>(segsum, (size_t)Nn * w);
    rago_fill0<<<ewg((size_t)Nn), 256, 0, stream>>>(segcnt, (size_t)Nn);
    rago_seg_accum<<<ewg((size_t)E * w), 256, 0, stream>>>(vals, ei1, segsum, segcnt, E, w);
    rago_seg_div<<<ewg((size_t)Nn * w), 256, 0, stream>>>(segsum, segcnt, out, Nn, w);
  };
  auto mlp6 = [&](const float* const* Wl, const float* const* bl,
                  const float* x0, int xw0, const float* x1, int xw1,
                  float* ta, float* tb, float* tc, float* out, int M, int lastN) {
    gemm(x0, xw0, nullptr, x1, xw1, nullptr, nullptr, 0, nullptr, Wl[0], bl[0], ta, M, HDIM, ACT_RELU);
    gemm(ta, HDIM, nullptr, nullptr, 0, nullptr, nullptr, 0, nullptr, Wl[1], bl[1], tb, M, HDIM, ACT_RELU);
    gemm(tb, HDIM, nullptr, nullptr, 0, nullptr, nullptr, 0, nullptr, Wl[2], bl[2], tc, M, HDIM, ACT_RELU);
    gemm(tc, HDIM, nullptr, x0, xw0, nullptr, x1, xw1, nullptr, Wl[3], bl[3], ta, M, HDIM, ACT_RELU);
    gemm(ta, HDIM, nullptr, nullptr, 0, nullptr, nullptr, 0, nullptr, Wl[4], bl[4], tb, M, HDIM, ACT_RELU);
    gemm(tb, HDIM, nullptr, nullptr, 0, nullptr, nullptr, 0, nullptr, Wl[5], bl[5], out, M, lastN, ACT_NONE);
  };
  auto gruStep = [&](const float* Wz, const float* bz, const float* Wr, const float* br,
                     const float* Wq, const float* bq,
                     float* h, const float* x0, const float* x1,
                     float* tz, float* tr, float* trh, float* tq, int M) {
    gemm(h, HDIM, nullptr, x0, HDIM, nullptr, x1, HDIM, nullptr, Wz, bz, tz, M, HDIM, ACT_SIG);
    gemm(h, HDIM, nullptr, x0, HDIM, nullptr, x1, HDIM, nullptr, Wr, br, tr, M, HDIM, ACT_SIG);
    rago_mul<<<ewg((size_t)M * HDIM), 256, 0, stream>>>(tr, h, trh, (size_t)M * HDIM);
    gemm(trh, HDIM, nullptr, x0, HDIM, nullptr, x1, HDIM, nullptr, Wq, bq, tq, M, HDIM, ACT_TANH);
    rago_gru_combine<<<ewg((size_t)M * HDIM), 256, 0, stream>>>(h, tz, tq, (size_t)M * HDIM);
  };
  auto graphCost = [&]() {  // ecost -> t0 (Ex9), nc -> n0 (Nn x 9)
    rago_fill0<<<ewg((size_t)Nn * 9), 256, 0, stream>>>(segsum, (size_t)Nn * 9);
    rago_fill0<<<ewg((size_t)Nn), 256, 0, stream>>>(segcnt, (size_t)Nn);
    rago_graph_cost<<<ewg((size_t)E), 256, 0, stream>>>(node_rot, edge_rot, edge_attr,
                                                        ei0, ei1, t0, segsum, segcnt, E);
    rago_seg_div<<<ewg((size_t)Nn * 9), 256, 0, stream>>>(segsum, segcnt, n0, Nn, 9);
  };

  // ---- init: node_rot = init_rot, edge_rot = 0 ----
  hipMemcpyAsync(node_rot, init_rot, (size_t)Nn * 9 * sizeof(float),
                 hipMemcpyDeviceToDevice, stream);
  rago_fill0<<<ewg((size_t)E * 6), 256, 0, stream>>>(edge_rot, (size_t)E * 6);

  // ---- initial mpnn_feat / mpnn_state convolutions ----
  gemm(node_rot, 9, ei0, node_rot, 9, ei1, edge_attr, 9, nullptr,
       MF_We, MF_be, edge_feat, E, HDIM, ACT_RELU);
  segmean(edge_feat, HDIM, n0);
  gemm(node_rot, 9, nullptr, n0, HDIM, nullptr, nullptr, 0, nullptr,
       MF_Wn, MF_bn, node_feat, Nn, HDIM, ACT_RELU);
  gemm(node_rot, 9, ei0, node_rot, 9, ei1, edge_attr, 9, nullptr,
       MS_We, MS_be, edge_state, E, HDIM, ACT_RELU);
  segmean(edge_state, HDIM, n0);
  gemm(node_rot, 9, nullptr, n0, HDIM, nullptr, nullptr, 0, nullptr,
       MS_Wn, MS_bn, node_state, Nn, HDIM, ACT_RELU);
  rago_tanh_ip<<<ewg((size_t)E * HDIM), 256, 0, stream>>>(edge_state, (size_t)E * HDIM);
  rago_tanh_ip<<<ewg((size_t)Nn * HDIM), 256, 0, stream>>>(node_state, (size_t)Nn * HDIM);

  float* rect = (float*)d_out;                 // (3, E, 6)
  float* est  = rect + (size_t)3 * E * 6;      // (12, Nn, 9)
  int esti = 0;

  for (int ig = 0; ig < 3; ++ig) {  // TG
    // ===== edge refinement (TE = 1) =====
    graphCost();
    gemm(n0, 9, ei0, n0, 9, ei1, t0, 9, nullptr, MEC_We[0], MEC_be[0], t1, E, HDIM, ACT_RELU);
    segmean(t1, HDIM, n1);
    gemm(n0, 9, nullptr, n1, HDIM, nullptr, nullptr, 0, nullptr, MEC_Wn[0], MEC_bn[0], n2, Nn, HDIM, ACT_RELU);
    gemm(n2, HDIM, ei0, n2, HDIM, ei1, t1, HDIM, nullptr, MEC_We[1], MEC_be[1], t2, E, HDIM, ACT_RELU);
    segmean(t2, HDIM, n1);
    gemm(n2, HDIM, nullptr, n1, HDIM, nullptr, nullptr, 0, nullptr, MEC_Wn[1], MEC_bn[1], n3, Nn, HDIM, ACT_RELU);
    gemm(n3, HDIM, ei0, n3, HDIM, ei1, t2, HDIM, nullptr, MEC_We[2], MEC_be[2], t3, E, HDIM, ACT_RELU); // ecf
    mlp6(EF_W, EF_b, t3, HDIM, edge_feat, HDIM, t0, t1, t2, t4, E, HDIM);
    gruStep(EG_Wz, EG_bz, EG_Wr, EG_br, EG_Wq, EG_bq,
            edge_state, t4, edge_feat, t0, t2, t3, t1, E);
    mlp6(EU_W, EU_b, edge_state, HDIM, nullptr, 0, t0, t1, t2, t3, E, 6);
    rago_edge_update<<<ewg((size_t)E * 6), 256, 0, stream>>>(
        edge_rot, t3, rect + (size_t)ig * E * 6, (size_t)E * 6);

    // ===== node refinement (TN = 4) =====
    for (int it = 0; it < 4; ++it) {
      graphCost();
      gemm(n0, 9, ei0, n0, 9, ei1, t0, 9, nullptr, MNC_We[0], MNC_be[0], t1, E, HDIM, ACT_RELU);
      segmean(t1, HDIM, n1);
      gemm(n0, 9, nullptr, n1, HDIM, nullptr, nullptr, 0, nullptr, MNC_Wn[0], MNC_bn[0], n2, Nn, HDIM, ACT_RELU);
      gemm(n2, HDIM, ei0, n2, HDIM, ei1, t1, HDIM, nullptr, MNC_We[1], MNC_be[1], t2, E, HDIM, ACT_RELU);
      segmean(t2, HDIM, n1);
      gemm(n2, HDIM, nullptr, n1, HDIM, nullptr, nullptr, 0, nullptr, MNC_Wn[1], MNC_bn[1], n3, Nn, HDIM, ACT_RELU);
      gemm(n3, HDIM, ei0, n3, HDIM, ei1, t2, HDIM, nullptr, MNC_We[2], MNC_be[2], t1, E, HDIM, ACT_RELU);
      segmean(t1, HDIM, n1);
      gemm(n3, HDIM, nullptr, n1, HDIM, nullptr, nullptr, 0, nullptr, MNC_Wn[2], MNC_bn[2], n0, Nn, HDIM, ACT_RELU); // ncf
      mlp6(NF_W, NF_b, n0, HDIM, node_feat, HDIM, n1, n2, n3, n4, Nn, HDIM);
      gruStep(NG_Wz, NG_bz, NG_Wr, NG_br, NG_Wq, NG_bq,
              node_state, n4, node_feat, n1, n2, n3, n0, Nn);
      mlp6(NU_W, NU_b, node_state, HDIM, nullptr, 0, n1, n2, n3, n0, Nn, 6); // d6
      rago_node_update<<<ewg((size_t)Nn), 256, 0, stream>>>(
          node_rot, n0, est + (size_t)esti * Nn * 9, Nn);
      ++esti;
    }
  }
}
